// MyModule_23046794510402
// MI455X (gfx1250) — 8-bit matrix operand order repaired and checked by reading, compiled, not run on hardware
//
#include <hip/hip_runtime.h>

typedef __attribute__((ext_vector_type(16))) int   v16i;
typedef __attribute__((ext_vector_type(8)))  float v8f;
typedef __attribute__((ext_vector_type(4)))  int   v4i;

#define K_DIM  8192
#define KB4    (K_DIM / 2)
#define NBLK   (K_DIM / 32)
#define N_DIM  8192
#define M_DIM  16
#define KSTEPS (K_DIM / 128)

__device__ __forceinline__ void mklut(float s, uint32_t* lo, uint32_t* hi) {
  const int a = (((int)((__float_as_uint(s) >> 23) & 0xFFu)) - 125) << 3;
  const uint32_t rep = (uint32_t)a * 0x01010101u;
  *lo = 0x2C282000u + (rep & 0xFFFFFF00u);
  *hi = 0x3C383430u + rep;
}

__device__ __forceinline__ uint32_t pack4(v4i q) {
  return (uint32_t)q[0] | ((uint32_t)q[1] << 8) |
         ((uint32_t)q[2] << 16) | ((uint32_t)q[3] << 24);
}

__device__ __forceinline__ void dq8(uint32_t packed, uint32_t lutLo, uint32_t lutHi,
                                    uint32_t* d0, uint32_t* d1) {
  const uint32_t hi = packed >> 4;
  const uint32_t nib0 = __builtin_amdgcn_perm(hi, packed, 0x05010400u);
  const uint32_t nib1 = __builtin_amdgcn_perm(hi, packed, 0x07030602u);
  *d0 = __builtin_amdgcn_perm(lutHi, lutLo, nib0 & 0x07070707u) |
        ((nib0 << 4) & 0x80808080u);
  *d1 = __builtin_amdgcn_perm(lutHi, lutLo, nib1 & 0x07070707u) |
        ((nib1 << 4) & 0x80808080u);
}

__global__ __launch_bounds__(256)
void mxfp4_wmma_gemm(const int* __restrict__ x, const float* __restrict__ xs,
                     const int* __restrict__ w, const float* __restrict__ ws,
                     float* __restrict__ out) {
  const int lane = threadIdx.x & 31;
  const int wave = threadIdx.x >> 5;
  const int ln = lane & 15;
  const int lh = lane >> 4;

  const int batch = blockIdx.x >> 6;
  const int ntile = ((blockIdx.x & 63) << 3) + wave;
  const int n0 = ntile << 4;

  const int*   xrow = x  + ((size_t)batch * M_DIM + ln) * KB4;
  const float* xsr  = xs + ((size_t)batch * M_DIM + ln) * NBLK;
  const int*   wrow = w  + ((size_t)batch * N_DIM + n0 + ln) * KB4;
  const float* wsr  = ws + ((size_t)batch * N_DIM + n0 + ln) * NBLK;

  v8f acc = {};

  for (int ks = 0; ks < KSTEPS; ++ks) {
    const int kb4  = ks << 6;
    const int blk0 = ks << 2;

    __builtin_prefetch(wrow + kb4 + 64 + (lh << 4), 0, 1);

    uint32_t alo[4], ahi[4], blo[4], bhi[4];
#pragma unroll
    for (int g = 0; g < 4; ++g) {
      mklut(xsr[blk0 + g], &alo[g], &ahi[g]);
      mklut(__builtin_nontemporal_load(wsr + blk0 + g), &blo[g], &bhi[g]);
    }

    v16i A;
#pragma unroll
    for (int p = 0; p < 8; ++p) {
      const int v = p << 1;
      const int k0 = ((v >> 2) << 5) + (((v >> 1) & 1) << 3) + (lh << 4);
      const v4i q = *(const v4i*)(xrow + kb4 + (k0 >> 1));
      uint32_t d0, d1;
      dq8(pack4(q), alo[v >> 2], ahi[v >> 2], &d0, &d1);
      A[v] = (int)d0;
      A[v + 1] = (int)d1;
    }

    v16i Bm;
#pragma unroll
    for (int g = 0; g < 4; ++g) {
      const int base = kb4 + ((g << 5) + (lh << 4) >> 1);
      const v4i q0 = __builtin_nontemporal_load((const v4i*)(wrow + base));
      const v4i q1 = __builtin_nontemporal_load((const v4i*)(wrow + base + 4));
      uint32_t d0, d1, d2, d3;
      dq8(pack4(q0), blo[g], bhi[g], &d0, &d1);
      dq8(pack4(q1), blo[g], bhi[g], &d2, &d3);
      Bm[4 * g] = (int)d0;
      Bm[4 * g + 1] = (int)d1;
      Bm[4 * g + 2] = (int)d2;
      Bm[4 * g + 3] = (int)d3;
    }

    acc = __builtin_amdgcn_wmma_f32_16x16x128_fp8_fp8(A, Bm, (short)0, acc,
                                                      false, false);
  }

  float* orow = out + (size_t)batch * M_DIM * N_DIM + n0 + ln;
#pragma unroll
  for (int j = 0; j < 8; ++j) {
    const int m = j + (lh << 3);
    __builtin_nontemporal_store(acc[j], orow + (size_t)m * N_DIM);
  }
}

extern "C" void kernel_launch(void* const* d_in, const int* in_sizes, int n_in,
                              void* d_out, int out_size, void* d_ws, size_t ws_size,
                              hipStream_t stream) {
  (void)in_sizes; (void)n_in; (void)out_size; (void)d_ws; (void)ws_size;
  const int*   x   = (const int*)d_in[0];
  const float* xsc = (const float*)d_in[1];
  const int*   w   = (const int*)d_in[2];
  const float* wsc = (const float*)d_in[3];
  float* out = (float*)d_out;

  dim3 grid(4 * (N_DIM / 128));
  dim3 block(256);
  hipLaunchKernelGGL(mxfp4_wmma_gemm, grid, block, 0, stream,
                     x, xsc, w, wsc, out);
}
